// AlphaEntmaxRouter_21234318311689
// MI455X (gfx1250) — matrix operand order repaired and checked by reading, not compiled, not run on hardware
//
#include <hip/hip_runtime.h>

// ---------------------------------------------------------------------------
// AlphaEntmaxRouter for MI455X (gfx1250, wave32, WMMA)
//   logits = x[32768,2048] @ W[64,2048]^T + b  ->  entmax_bisect(alpha=1.5)
// Fused: GEMM (split-bf16 WMMA, fp32 accum) + entmax, logits never hit HBM.
// ---------------------------------------------------------------------------

typedef __attribute__((ext_vector_type(16))) __bf16 v16bf;
typedef __attribute__((ext_vector_type(8)))  __bf16 v8bf;
typedef __attribute__((ext_vector_type(8)))  float  v8f;
typedef __attribute__((ext_vector_type(4)))  float  f32x4;
typedef __attribute__((ext_vector_type(4)))  unsigned int u32x4;

#define N_EMBED 2048
#define N_EXP   64
#define N_TOK   32768
#define ROWS_PER_BLOCK 128   // 8 waves * 16 rows
#define THREADS 256

// Split a float8 into bf16 hi (RNE, native v_cvt_pk_bf16_f32) + bf16 lo
// residual. hf is rebuilt from the packed hi with shifts/ands (dual-issue)
// instead of re-converting.
__device__ __forceinline__ void split8(v8f f, v8bf& hi, v8bf& lo) {
  hi = __builtin_convertvector(f, v8bf);            // 4x v_cvt_pk_bf16_f32
  u32x4 u = __builtin_bit_cast(u32x4, hi);
  v8f hf;
  #pragma unroll
  for (int i = 0; i < 4; ++i) {
    hf[2 * i]     = __builtin_bit_cast(float, u[i] << 16);
    hf[2 * i + 1] = __builtin_bit_cast(float, u[i] & 0xFFFF0000u);
  }
  lo = __builtin_convertvector(f - hf, v8bf);       // 4x pk_add + 4x cvt_pk
}

// --- pre-pass: split W (fp32) into bf16 hi + lo residual -------------------
__global__ __launch_bounds__(THREADS) void wsplit_kernel(const float* __restrict__ W,
                                                         __bf16* __restrict__ Whi,
                                                         __bf16* __restrict__ Wlo) {
  int i = blockIdx.x * blockDim.x + threadIdx.x;
  if (i < N_EXP * N_EMBED) {
    float w = W[i];
    __bf16 h = (__bf16)w;
    Whi[i] = h;
    Wlo[i] = (__bf16)(w - (float)h);
  }
}

// --- fused GEMM + entmax ----------------------------------------------------
__global__ __launch_bounds__(THREADS) void entmax_router_kernel(
    const float* __restrict__ x,
    const __bf16* __restrict__ Whi,
    const __bf16* __restrict__ Wlo,
    const float* __restrict__ bias,
    float* __restrict__ out)
{
  __shared__ float lds[ROWS_PER_BLOCK * N_EXP];   // 32 KB, XOR-swizzled

  const int tid  = threadIdx.x;
  const int lane = tid & 31;
  const int wv   = tid >> 5;          // wave id 0..7
  const int hlf  = lane >> 4;         // which 16-lane half
  const int l16  = lane & 15;
  const size_t blockRow = (size_t)blockIdx.x * ROWS_PER_BLOCK;

  // ---- GEMM phase: each wave computes rows [wv*16, wv*16+16) x all 64 experts
  v8f acc[4];
  #pragma unroll
  for (int j = 0; j < 4; ++j) acc[j] = (v8f){0.f,0.f,0.f,0.f,0.f,0.f,0.f,0.f};

  const size_t rowA = blockRow + (size_t)wv * 16 + l16;   // A row = token (M = l16)
  const float* pA = x + rowA * N_EMBED;
  const int koffA = hlf * 16;   // A k order (contiguous, matches B): lanes 0-15 K=0..15 ; 16-31 K=16..31
  const int koffB = hlf * 16;   // B k order (contiguous, matches A): lanes 0-15 K=0..15 ; 16-31 K=16..31

  for (int k = 0; k < N_EMBED; k += 32) {
    // prefetch the A stream ~4 k-steps ahead (x is a pure HBM stream)
    __builtin_prefetch(pA + k + koffA + 128, 0, 3);

    // load 16 fp32 A values per lane (two 32B chunks), split into bf16 hi/lo
    v8f f0 = *(const v8f*)(pA + k + koffA);
    v8f f1 = *(const v8f*)(pA + k + koffA + 8);
    v8bf h0, l0, h1, l1;
    split8(f0, h0, l0);
    split8(f1, h1, l1);
    v16bf ah = __builtin_shufflevector(h0, h1, 0,1,2,3,4,5,6,7,8,9,10,11,12,13,14,15);
    v16bf al = __builtin_shufflevector(l0, l1, 0,1,2,3,4,5,6,7,8,9,10,11,12,13,14,15);

    // keep all 8 B operands resident, then issue product-major WMMAs so each
    // accumulator has 4 independent WMMAs between reuses (hazard-free)
    v16bf bh[4], bl[4];
    #pragma unroll
    for (int j = 0; j < 4; ++j) {
      const size_t wrow = (size_t)(j * 16 + l16) * N_EMBED + (size_t)(k + koffB);
      bh[j] = *(const v16bf*)(Whi + wrow);
      bl[j] = *(const v16bf*)(Wlo + wrow);
    }
    #pragma unroll
    for (int j = 0; j < 4; ++j)
      acc[j] = __builtin_amdgcn_wmma_f32_16x16x32_bf16(false, ah, false, bh[j], (short)0, acc[j], false, false);
    #pragma unroll
    for (int j = 0; j < 4; ++j)
      acc[j] = __builtin_amdgcn_wmma_f32_16x16x32_bf16(false, ah, false, bl[j], (short)0, acc[j], false, false);
    #pragma unroll
    for (int j = 0; j < 4; ++j)
      acc[j] = __builtin_amdgcn_wmma_f32_16x16x32_bf16(false, al, false, bh[j], (short)0, acc[j], false, false);
  }

  // ---- dump logits to LDS (C layout: VGPR r -> M=r(+8 for hi half), N=l16)
  // XOR swizzle keeps 4-float groups contiguous -> entmax can use ds_load_b128
  {
    const int rowb = wv * 16 + hlf * 8;
    #pragma unroll
    for (int j = 0; j < 4; ++j) {
      const int col = j * 16 + l16;
      #pragma unroll
      for (int r = 0; r < 8; ++r) {
        const int row = rowb + r;
        lds[row * N_EXP + (col ^ ((row & 15) << 2))] = acc[j][r];
      }
    }
  }
  __syncthreads();

  // ---- entmax phase: threads 0..127 each own one token row ----------------
  if (tid < ROWS_PER_BLOCK) {
    const int trow = tid;
    const int xsw = (trow & 15) << 2;
    float X[N_EXP];
    #pragma unroll
    for (int g = 0; g < N_EXP / 4; ++g) {
      f32x4 v4 = *(const f32x4*)&lds[trow * N_EXP + ((g * 4) ^ xsw)];
      #pragma unroll
      for (int e = 0; e < 4; ++e)
        X[g * 4 + e] = (v4[e] + bias[g * 4 + e]) * 0.5f;  // (logits/T)*(alpha-1)
    }
    float maxv = X[0];
    #pragma unroll
    for (int i = 1; i < N_EXP; ++i) maxv = fmaxf(maxv, X[i]);

    float tau_lo = maxv - 1.0f;
    float tau_hi = maxv - 0.000244140625f;      // (1/64)^(1/(alpha-1)) = (1/64)^2
    float dm     = tau_hi - tau_lo;

    float f_lo = -1.0f;                          // sum p(tau_lo) - 1
    #pragma unroll
    for (int i = 0; i < N_EXP; ++i) {
      float t = fmaxf(X[i] - tau_lo, 0.0f);
      f_lo += t * t;                             // inv = 1/(alpha-1) = 2 -> square
    }

    float tau_m = tau_lo;
    for (int it = 0; it < 25; ++it) {
      dm *= 0.5f;
      tau_m = tau_lo + dm;
      float f_m = -1.0f;
      #pragma unroll
      for (int i = 0; i < N_EXP; ++i) {
        float t = fmaxf(X[i] - tau_m, 0.0f);
        f_m += t * t;
      }
      if (f_m * f_lo >= 0.0f) tau_lo = tau_m;    // f_lo intentionally fixed (matches ref)
    }

    // final p uses LAST tau_m (exactly like tau_ms[-1] in the reference)
    float p[N_EXP];
    float s = 0.0f;
    #pragma unroll
    for (int i = 0; i < N_EXP; ++i) {
      float t = fmaxf(X[i] - tau_m, 0.0f);
      p[i] = t * t;
      s += p[i];
    }
    const float invs = 1.0f / s;
    float* po = out + (blockRow + (size_t)trow) * N_EXP;
    #pragma unroll
    for (int i = 0; i < N_EXP; i += 4) {
      f32x4 v = { p[i] * invs, p[i + 1] * invs, p[i + 2] * invs, p[i + 3] * invs };
      *(f32x4*)(po + i) = v;
    }
  }
}

// ---------------------------------------------------------------------------
extern "C" void kernel_launch(void* const* d_in, const int* in_sizes, int n_in,
                              void* d_out, int out_size, void* d_ws, size_t ws_size,
                              hipStream_t stream) {
  (void)in_sizes; (void)n_in; (void)out_size; (void)ws_size;
  const float* x = (const float*)d_in[0];   // [32768, 2048] fp32
  const float* W = (const float*)d_in[1];   // [64, 2048]    fp32
  const float* b = (const float*)d_in[2];   // [64]          fp32
  float* out = (float*)d_out;               // [32768, 64]   fp32

  __bf16* Whi = (__bf16*)d_ws;              // 256 KB
  __bf16* Wlo = Whi + (size_t)N_EXP * N_EMBED;  // +256 KB  (needs 512 KB ws)

  wsplit_kernel<<<(N_EXP * N_EMBED + THREADS - 1) / THREADS, THREADS, 0, stream>>>(W, Whi, Wlo);
  entmax_router_kernel<<<N_TOK / ROWS_PER_BLOCK, THREADS, 0, stream>>>(x, Whi, Wlo, b, out);
}
